// GATLayer_33947421507722
// MI455X (gfx1250) — matrix operand order repaired and checked by reading, not compiled, not run on hardware
//
#include <hip/hip_runtime.h>
#include <hip/hip_bf16.h>

typedef _Float16 half_t;
typedef __attribute__((ext_vector_type(16))) _Float16 v16h;
typedef __attribute__((ext_vector_type(8)))  _Float16 v8h;
typedef __attribute__((ext_vector_type(8)))  float    v8f;
typedef __attribute__((ext_vector_type(4)))  float    v4f;
typedef __attribute__((ext_vector_type(4)))  int      v4i;

#define B_    8
#define N_    2048
#define FIN   128
#define FOUT  64
#define ALPHA 0.2f

// --- CDNA5 async global->LDS path (gate on builtin availability) -----------
#if __has_builtin(__builtin_amdgcn_global_load_async_to_lds_b128) && \
    __has_builtin(__builtin_amdgcn_s_wait_asynccnt)
#define HAVE_ASYNC 1
#else
#define HAVE_ASYNC 0
#endif

typedef __attribute__((address_space(1))) v4i gas_v4i;   // global-AS int4
typedef __attribute__((address_space(3))) v4i las_v4i;   // LDS-AS int4

__device__ __forceinline__ void stage16B(const int* g, int* l) {
#if HAVE_ASYNC
    __builtin_amdgcn_global_load_async_to_lds_b128(
        (gas_v4i*)(uintptr_t)g,
        (las_v4i*)(unsigned int)(uintptr_t)l,
        0, 0);
#else
    *(v4i*)l = *(const v4i*)g;     // sync fallback: global_load + ds_store
#endif
}

__device__ __forceinline__ void wait_async_le(int n) {
#if HAVE_ASYNC
    if (n == 0)       __builtin_amdgcn_s_wait_asynccnt(0);
    else if (n == 4)  __builtin_amdgcn_s_wait_asynccnt(4);
    else if (n == 8)  __builtin_amdgcn_s_wait_asynccnt(8);
    else              __builtin_amdgcn_s_wait_asynccnt(12);
#endif
    asm volatile("" ::: "memory");  // keep ds reads behind the wait
}

// ---------------------------------------------------------------------------
// Kernel 2 (defined first so disasm snippet shows the async pipeline):
// fused masked-softmax attention + P@Wh, single pass over adj.
// adj tiles (16x32 ints, 2 KB) are staged into LDS with async global->LDS
// copies, 4-deep pipelined on ASYNCcnt. One wave per (batch, 16-row) tile.
// ---------------------------------------------------------------------------
#define PITCH 36   // LDS tile row pitch in dwords (16B-aligned, de-conflicted)

__global__ __launch_bounds__(32) void gat_attn_kernel(
    const int* __restrict__ adj, const half_t* __restrict__ whfrag,
    const float* __restrict__ s1, const float* __restrict__ s2,
    const float* __restrict__ s2max, float* __restrict__ out)
{
    __shared__ int lds_adj[4][16 * PITCH];        // 4 x 2.25 KB buffers

    const int lane    = threadIdx.x;
    const int col     = lane & 15;
    const int half_id = lane >> 4;
    const int b       = blockIdx.x >> 7;
    const int it      = blockIdx.x & 127;
    const int i       = it * 16 + col;            // A-matrix row for this lane

    const float s1v = s1[b * N_ + i];
    const float mx  = s1v + s2max[b];
    const float m_i = mx >= 0.f ? mx : ALPHA * mx;

    const int*   adjb = adj + (size_t)b * N_ * N_;
    const float* s2b  = s2 + b * N_;

    v8f   acc[4] = {};
    float dpart  = 0.f;

    // coalesced stage of one 16x32 adj tile: 4 async b128 ops, 8 lanes/row
    const int rsub = lane >> 3;                   // 0..3
    const int cq   = (lane & 7) * 4;              // dword column
    auto issue = [&](int js, int buf) {
        #pragma unroll
        for (int k = 0; k < 4; ++k) {
            const int row = k * 4 + rsub;
            const int* g  = adjb + (size_t)(it * 16 + row) * N_ + js * 32 + cq;
            stage16B(g, &lds_adj[buf][row * PITCH + cq]);
        }
    };

    auto mkp = [&](int av, float s2v) -> half_t {
        float ee = s1v + s2v;
        ee = ee >= 0.f ? ee : ALPHA * ee;
        const float p = (av > 0) ? __expf(ee - m_i) : 0.f;
        dpart += p;
        return (half_t)p;
    };

    auto consume = [&](int js, int buf) {
        const int* tile = &lds_adj[buf][0];
        const int  base = col * PITCH + half_id * 8;       // K runs per A layout
        const v4i aA0 = *(const v4i*)(tile + base);
        const v4i aA1 = *(const v4i*)(tile + base + 4);
        const v4i aB0 = *(const v4i*)(tile + base + 16);
        const v4i aB1 = *(const v4i*)(tile + base + 20);

        const int jA = js * 32 + half_id * 8;
        const int jB = jA + 16;
        const v4f s2A0 = *(const v4f*)(s2b + jA);
        const v4f s2A1 = *(const v4f*)(s2b + jA + 4);
        const v4f s2B0 = *(const v4f*)(s2b + jB);
        const v4f s2B1 = *(const v4f*)(s2b + jB + 4);

        v16h pfrag;
        #pragma unroll
        for (int e = 0; e < 4; ++e) pfrag[e]      = mkp(aA0[e], s2A0[e]);
        #pragma unroll
        for (int e = 0; e < 4; ++e) pfrag[4 + e]  = mkp(aA1[e], s2A1[e]);
        #pragma unroll
        for (int e = 0; e < 4; ++e) pfrag[8 + e]  = mkp(aB0[e], s2B0[e]);
        #pragma unroll
        for (int e = 0; e < 4; ++e) pfrag[12 + e] = mkp(aB1[e], s2B1[e]);

        const half_t* fb = whfrag + (size_t)((b * 64 + js) * 4) * 32 * 16;
        #pragma unroll
        for (int c = 0; c < 4; ++c) {
            const v8h* src = (const v8h*)(fb + (size_t)(c * 32 + lane) * 16);
            const v8h lo = src[0];
            const v8h hi = src[1];
            v16h bfrag;
            #pragma unroll
            for (int e = 0; e < 8; ++e) { bfrag[e] = lo[e]; bfrag[8 + e] = hi[e]; }
            acc[c] = __builtin_amdgcn_wmma_f32_16x16x32_f16(
                false, pfrag, false, bfrag, (short)0, acc[c], false, false);
        }
    };

    // 4-deep async pipeline over 64 j-tiles
    issue(0, 0); issue(1, 1); issue(2, 2);
    for (int js = 0; js < 61; ++js) {
        issue(js + 3, (js + 3) & 3);
        wait_async_le(12);                 // 3 tiles in flight -> tile js ready
        consume(js, js & 3);
    }
    wait_async_le(8);  consume(61, 61 & 3);
    wait_async_le(4);  consume(62, 62 & 3);
    wait_async_le(0);  consume(63, 63 & 3);

    // Row denominators: after pair-reduce, lane m holds total for row m.
    const float dtot = dpart + __shfl_xor(dpart, 16, 32);
    #pragma unroll
    for (int v = 0; v < 8; ++v) {
        const int   row = 8 * half_id + v;
        const float dv  = __shfl(dtot, row, 32);
        const float inv = 1.0f / dv;
        #pragma unroll
        for (int c = 0; c < 4; ++c) {
            const float x = acc[c][v] * inv;
            const float r = x > 0.f ? x : (__expf(x) - 1.0f);   // elu
            out[(size_t)(b * N_ + it * 16 + row) * FOUT + c * 16 + col] = r;
        }
    }
}

// ---------------------------------------------------------------------------
// Kernel 0: pre-swizzle W (128x64 f32) into B-fragment f16 layout, once.
// frag[(ks*4+c)*32 + lane][e] = W[ks*32 + (e<8 ? half*8 + e : 16 + half*8 + e-8)][c*16 + col]
// ---------------------------------------------------------------------------
__global__ __launch_bounds__(32) void gat_wfrag_kernel(
    const float* __restrict__ W, half_t* __restrict__ wfrag)
{
    const int lane    = threadIdx.x;
    const int col     = lane & 15;
    const int half_id = lane >> 4;
    #pragma unroll
    for (int ks = 0; ks < 4; ++ks) {
        #pragma unroll
        for (int c = 0; c < 4; ++c) {
            v8h lo, hi;
            #pragma unroll
            for (int e = 0; e < 8; ++e) {
                lo[e] = (half_t)W[(ks * 32 + half_id * 8 + e)      * FOUT + c * 16 + col];
                hi[e] = (half_t)W[(ks * 32 + 16 + half_id * 8 + e) * FOUT + c * 16 + col];
            }
            v8h* dst = (v8h*)(wfrag + (size_t)((ks * 4 + c) * 32 + lane) * 16);
            dst[0] = lo;
            dst[1] = hi;
        }
    }
}

// ---------------------------------------------------------------------------
// Kernel 1: Wh = h @ W (f16 WMMA, f32 accum), s1 = Wh@a1, s2 = Wh@a2,
// and emit Wh in B-fragment-swizzled f16 layout for the attention GEMM.
// ---------------------------------------------------------------------------
__global__ __launch_bounds__(32) void gat_wh_kernel(
    const float* __restrict__ h, const half_t* __restrict__ wfrag,
    const float* __restrict__ a, half_t* __restrict__ whfrag,
    float* __restrict__ s1, float* __restrict__ s2)
{
    __shared__ half_t lds_wh[32 * FOUT];
    const int lane    = threadIdx.x;
    const int col     = lane & 15;
    const int half_id = lane >> 4;
    const int tile    = blockIdx.x;
    const int row0    = tile * 32;

    float a1c[4], a2c[4];
    #pragma unroll
    for (int c = 0; c < 4; ++c) {
        a1c[c] = a[c * 16 + col];
        a2c[c] = a[FOUT + c * 16 + col];
    }

    #pragma unroll
    for (int t = 0; t < 2; ++t) {
        const int r = row0 + t * 16 + col;
        v8f acc[4] = {};
        #pragma unroll
        for (int ks = 0; ks < 4; ++ks) {
            // A fragment: vectorized b128 loads of two 8-float runs, cvt to f16
            const float* hrow = h + (size_t)r * FIN + ks * 32 + half_id * 8;
            const v4f h0 = *(const v4f*)(hrow);
            const v4f h1 = *(const v4f*)(hrow + 4);
            const v4f h2 = *(const v4f*)(hrow + 16);
            const v4f h3 = *(const v4f*)(hrow + 20);
            v16h afrag;
            #pragma unroll
            for (int e = 0; e < 4; ++e) {
                afrag[e]      = (half_t)h0[e];
                afrag[4 + e]  = (half_t)h1[e];
                afrag[8 + e]  = (half_t)h2[e];
                afrag[12 + e] = (half_t)h3[e];
            }
            // B fragments: contiguous pre-swizzled loads (2 x b128 each)
            #pragma unroll
            for (int c = 0; c < 4; ++c) {
                const v8h* wsrc = (const v8h*)(wfrag + (size_t)((ks * 4 + c) * 32 + lane) * 16);
                const v8h lo = wsrc[0];
                const v8h hi = wsrc[1];
                v16h bfrag;
                #pragma unroll
                for (int e = 0; e < 8; ++e) { bfrag[e] = lo[e]; bfrag[8 + e] = hi[e]; }
                acc[c] = __builtin_amdgcn_wmma_f32_16x16x32_f16(
                    false, afrag, false, bfrag, (short)0, acc[c], false, false);
            }
        }
        #pragma unroll
        for (int v = 0; v < 8; ++v) {
            float t1 = 0.f, t2 = 0.f;
            #pragma unroll
            for (int c = 0; c < 4; ++c) { t1 += acc[c][v] * a1c[c]; t2 += acc[c][v] * a2c[c]; }
            #pragma unroll
            for (int m = 1; m < 16; m <<= 1) {
                t1 += __shfl_xor(t1, m, 32);
                t2 += __shfl_xor(t2, m, 32);
            }
            if (col == 0) {
                const int grow = row0 + t * 16 + 8 * half_id + v;
                s1[grow] = t1;
                s2[grow] = t2;
            }
            #pragma unroll
            for (int c = 0; c < 4; ++c)
                lds_wh[(t * 16 + 8 * half_id + v) * FOUT + c * 16 + col] = (half_t)acc[c][v];
        }
    }
    __syncthreads();
    #pragma unroll
    for (int c = 0; c < 4; ++c) {
        v8h lo, hi;
        #pragma unroll
        for (int e = 0; e < 8; ++e) {
            lo[e] = lds_wh[(half_id * 8 + e)      * FOUT + c * 16 + col];
            hi[e] = lds_wh[(16 + half_id * 8 + e) * FOUT + c * 16 + col];
        }
        v8h* dst = (v8h*)(whfrag + (size_t)((tile * 4 + c) * 32 + lane) * 16);
        dst[0] = lo;
        dst[1] = hi;
    }
}

// ---------------------------------------------------------------------------
// Kernel 1b: per-batch max of s2 (monotone softmax shift)
// ---------------------------------------------------------------------------
__global__ __launch_bounds__(256) void gat_s2max_kernel(
    const float* __restrict__ s2, float* __restrict__ s2max)
{
    __shared__ float red[256];
    const int b = blockIdx.x;
    float m = -3.4e38f;
    for (int n = threadIdx.x; n < N_; n += 256)
        m = fmaxf(m, s2[b * N_ + n]);
    red[threadIdx.x] = m;
    __syncthreads();
    for (int s = 128; s > 0; s >>= 1) {
        if (threadIdx.x < s) red[threadIdx.x] = fmaxf(red[threadIdx.x], red[threadIdx.x + s]);
        __syncthreads();
    }
    if (threadIdx.x == 0) s2max[b] = red[0];
}

// ---------------------------------------------------------------------------
// Launch
// ---------------------------------------------------------------------------
extern "C" void kernel_launch(void* const* d_in, const int* in_sizes, int n_in,
                              void* d_out, int out_size, void* d_ws, size_t ws_size,
                              hipStream_t stream) {
    const float* h   = (const float*)d_in[0];
    const float* W   = (const float*)d_in[1];
    const float* a   = (const float*)d_in[2];
    const int*   adj = (const int*)d_in[3];
    float*       out = (float*)d_out;

    char*   ws     = (char*)d_ws;
    half_t* whfrag = (half_t*)ws;                                 // 2 MB
    float*  s1     = (float*)(ws + (2u << 20));                   // 64 KB
    float*  s2     = (float*)(ws + (2u << 20) + (64u << 10));     // 64 KB
    float*  s2m    = (float*)(ws + (2u << 20) + (128u << 10));    // 32 B
    half_t* wfrag  = (half_t*)(ws + (2u << 20) + (132u << 10));   // 16 KB

    gat_wfrag_kernel<<<1, 32, 0, stream>>>(W, wfrag);
    gat_wh_kernel<<<(B_ * N_) / 32, 32, 0, stream>>>(h, wfrag, a, whfrag, s1, s2);
    gat_s2max_kernel<<<B_, 256, 0, stream>>>(s2, s2m);
    gat_attn_kernel<<<B_ * (N_ / 16), 32, 0, stream>>>(adj, whfrag, s1, s2, s2m, out);
}
